// CNN_29609504539560
// MI455X (gfx1250) — matrix operand order repaired and checked by reading, not compiled, not run on hardware
//
#include <hip/hip_runtime.h>
#include <hip/hip_bf16.h>

typedef __attribute__((ext_vector_type(16))) _Float16 v16h;
typedef __attribute__((ext_vector_type(8)))  float    v8f;

// =====================================================================
// Weight prep: pack (w . radial_basis) into the V_WMMA_F32_16X16X32_F16
// B operand in contiguous K order (matches the A side). K-space = Cin*7*7*8 (kx padded to 8; slot kx==7,
// rows beyond Cin*49, and K-chunks padded to a multiple of 8 get zeros).
// Layout per 32-K chunk:
//   lanes 0-15 (N=lane): VGPR v -> K = 2v, 2v+1
//   lanes 16-31 (N=lane-16): VGPR v -> K = 16+2v, 17+2v
// Stored flat as wB[((nt*NCH + chunk)*32 + lane)*8 + v]  (u32 = 2 x f16).
// =====================================================================
__global__ void prep_wB_k(const float* __restrict__ w, unsigned* __restrict__ wB,
                          int Cout, int Cin, int NT, int NCH)
{
  const long total = (long)NT * NCH * 256;
  for (long idx = (long)blockIdx.x * blockDim.x + threadIdx.x; idx < total;
       idx += (long)gridDim.x * blockDim.x) {
    const int  slot  = (int)(idx & 7);
    const int  lane  = (int)((idx >> 3) & 31);
    const long rest  = idx >> 8;
    const int  chunk = (int)(rest % NCH);
    const int  nt    = (int)(rest / NCH);
    const int  cout  = nt * 16 + (lane & 15);
    unsigned packed = 0u;
    if (cout < Cout) {
      const int kbase = chunk * 32 + ((lane & 16) ? 16 : 0) + 2 * slot;
      unsigned short us[2] = {0, 0};
      for (int t = 0; t < 2; ++t) {
        const int k   = kbase + t;
        const int kx  = k & 7;
        const int row = k >> 3;
        const int c   = row / 49;
        float val = 0.f;
        if (kx < 7 && c < Cin) {
          const int kz = (row % 49) / 7;
          const int ky = row % 7;
          // radial shell basis: centers {0,1.5,3}, sigma 1.5
          const float dz = (float)(kz - 3), dy = (float)(ky - 3), dx = (float)(kx - 3);
          const float dist = sqrtf(dz * dz + dy * dy + dx * dx);
          #pragma unroll
          for (int r = 0; r < 3; ++r) {
            const float d = dist - 1.5f * r;
            val += w[((long)cout * Cin + c) * 3 + r] * expf(-d * d * (1.f / 4.5f));
          }
        }
        _Float16 hv = (_Float16)val;
        us[t] = *(unsigned short*)&hv;
      }
      packed = (unsigned)us[0] | ((unsigned)us[1] << 16);
    }
    wB[idx] = packed;
  }
}

// per-K-row A-side address offsets: c*Dp^3 + kz*Dp^2 + ky*Dp, -1 = zero row
__global__ void prep_rows_k(int* __restrict__ rowTab, int Cin, int Dp, int nrows)
{
  const int i = blockIdx.x * blockDim.x + threadIdx.x;
  if (i < nrows) {
    const int c = i / 49, rr = i % 49, kz = rr / 7, ky = rr % 7;
    rowTab[i] = (c < Cin) ? (int)((((long)c * Dp + kz) * Dp + ky) * Dp) : -1;
  }
}

// ============== layer-0 input: f32 -> f16 with 3-voxel halo ==============
__global__ void pad_input_k(const float* __restrict__ x, _Float16* __restrict__ xp,
                            int Bn, int C, int Din, int Dp)
{
  const long vol = (long)Dp * Dp * Dp;
  const long total = (long)Bn * C * vol;
  for (long idx = (long)blockIdx.x * blockDim.x + threadIdx.x; idx < total;
       idx += (long)gridDim.x * blockDim.x) {
    int xpx = (int)(idx % Dp); long t = idx / Dp;
    int ypx = (int)(t % Dp);   t /= Dp;
    int zpx = (int)(t % Dp);   t /= Dp;
    int c   = (int)(t % C);
    int b   = (int)(t / C);
    const int xi = xpx - 3, yi = ypx - 3, zi = zpx - 3;
    float v = 0.f;
    if (xi >= 0 && xi < Din && yi >= 0 && yi < Din && zi >= 0 && zi < Din)
      v = x[(((long)(b * C + c) * Din + zi) * Din + yi) * Din + xi];
    xp[idx] = (_Float16)v;
  }
}

// =====================================================================
// Implicit-GEMM conv. Block = 8 waves sharing (b, cout-tile); wave w owns
// 16-voxel row mtile = blk*8+w. B tiles (8 K-chunks = 8KB) are DMA'd into
// LDS with GLOBAL_LOAD_ASYNC_TO_LDS_B128, double-buffered on ASYNCcnt.
// =====================================================================
__global__ __launch_bounds__(256) void conv_wmma_k(
    const _Float16* __restrict__ xpad, const unsigned* __restrict__ wB,
    const int* __restrict__ rowTab, float* __restrict__ out,
    int Cin, int Cout, int NT, int NCH, int Dp, int stride)
{
  __shared__ unsigned Bs[2][2048];   // 2 x 8KB double buffer (8 chunks each)
  __shared__ int      rowS[5760];    // per-row A offsets (NCH*4 <= 5760)

  const int tid  = threadIdx.x;
  const int lane = tid & 31;
  const int wave = tid >> 5;
  const int bt   = blockIdx.x >> 5;          // (b*NT + nt)
  const int b    = bt / NT;
  const int nt   = bt - b * NT;
  const int mtile = ((blockIdx.x & 31) << 3) | wave;
  const int zo = mtile >> 4, yo = mtile & 15;
  const int mlane = lane & 15, halfsel = lane >> 4;

  const int nrows = NCH << 2;
  for (int i = tid; i < nrows; i += 256) rowS[i] = rowTab[i];

  const long plane2 = (long)Dp * Dp;
  const long laneBase = ((long)b * Cin) * plane2 * Dp
                      + (long)(stride * zo) * plane2
                      + (long)(stride * yo) * Dp
                      + (long)(stride * mlane);

  const unsigned* gB = wB + (long)nt * NCH * 256;       // uniform (SGPR pair)
  const unsigned  bs0 = (unsigned)(uintptr_t)&Bs[0][0]; // LDS byte offset

  // prologue: group 0 -> buffer 0 (32 B per thread)
  {
    const unsigned voff = (unsigned)tid * 32u;
    const unsigned la   = bs0 + (unsigned)tid * 32u;
    asm volatile("global_load_async_to_lds_b128 %0, %1, %2"
                 :: "v"(la), "v"(voff), "s"(gB) : "memory");
    asm volatile("global_load_async_to_lds_b128 %0, %1, %2"
                 :: "v"(la + 16u), "v"(voff + 16u), "s"(gB) : "memory");
  }

  union AU { v16h v; unsigned u[8]; };
  union BU { v16h v; uint4 q[2]; };
  v8f acc = {};
  const int ngroups = NCH >> 3;

  for (int g = 0; g < ngroups; ++g) {
    if (g + 1 < ngroups) {
      const unsigned voff = (unsigned)(g + 1) * 8192u + (unsigned)tid * 32u;
      const unsigned la   = bs0 + (unsigned)((g + 1) & 1) * 8192u + (unsigned)tid * 32u;
      asm volatile("global_load_async_to_lds_b128 %0, %1, %2"
                   :: "v"(la), "v"(voff), "s"(gB) : "memory");
      asm volatile("global_load_async_to_lds_b128 %0, %1, %2"
                   :: "v"(la + 16u), "v"(voff + 16u), "s"(gB) : "memory");
      // 2 newly issued stay in flight; group g (issued earlier) must land
      asm volatile("s_wait_asynccnt 2" ::: "memory");
    } else {
      asm volatile("s_wait_asynccnt 0" ::: "memory");
    }
    __syncthreads();

    const unsigned* bs = &Bs[g & 1][0];
    #pragma unroll
    for (int ci = 0; ci < 8; ++ci) {
      const int ch = (g << 3) + ci;
      BU bm;
      const uint4* bp = (const uint4*)(bs + ci * 256 + lane * 8);
      bm.q[0] = bp[0];
      bm.q[1] = bp[1];
      AU am;
      #pragma unroll
      for (int j = 0; j < 2; ++j) {
        const int off = rowS[(ch << 2) + (halfsel << 1) + j];
        if (off >= 0) {
          __builtin_memcpy(&am.u[4 * j], xpad + laneBase + off, 16); // 8 halves
        } else {
          am.u[4*j+0] = 0u; am.u[4*j+1] = 0u; am.u[4*j+2] = 0u; am.u[4*j+3] = 0u;
        }
      }
      acc = __builtin_amdgcn_wmma_f32_16x16x32_f16(
          /*neg_a*/false, am.v, /*neg_b*/false, bm.v,
          /*c_mod*/(short)0, acc, /*reuse_a*/false, /*reuse_b*/false);
    }
    __syncthreads();   // everyone done reading buf (g&1) before it refills
  }

  const int cout = nt * 16 + (lane & 15);
  if (cout < Cout) {
    float* o = out + ((long)b * Cout + cout) * 4096 + mtile * 16 + halfsel * 8;
    #pragma unroll
    for (int r = 0; r < 8; ++r) o[r] = acc[r];   // M = r + 8*halfsel
  }
}

// ======================= small utility kernels =======================
__global__ void zero_k(float* p, int n)
{
  int i = blockIdx.x * blockDim.x + threadIdx.x;
  if (i < n) p[i] = 0.f;
}

// per-(b,c) block: sum of squares over 4096 voxels, atomic into field slot
__global__ void norm_reduce_k(const float* __restrict__ y, float* __restrict__ norm2,
                              int Cout, int n1, int n3)
{
  const int bc = blockIdx.x;
  const int c  = bc % Cout;
  const float* p = y + (long)bc * 4096;
  float s = 0.f;
  for (int i = threadIdx.x; i < 4096; i += blockDim.x) { float v = p[i]; s += v * v; }
  __shared__ float sm[256];
  sm[threadIdx.x] = s; __syncthreads();
  for (int o = 128; o > 0; o >>= 1) {
    if (threadIdx.x < o) sm[threadIdx.x] += sm[threadIdx.x + o];
    __syncthreads();
  }
  if (threadIdx.x == 0) {
    const int f = (c < n1) ? c
                : (c < n1 + 3 * n3) ? n1 + (c - n1) / 3
                : n1 + n3 + (c - n1 - 3 * n3) / 5;
    atomicAdd(&norm2[f], sm[0]);
  }
}

// normalize + bias/ReLU(scalars) + l=1 tensor-product concat + halo-pad -> f16
__global__ void expand_pad_k(const float* __restrict__ y, const float* __restrict__ norm2,
                             const float* __restrict__ bias, _Float16* __restrict__ xp,
                             int Bn, int Cout, int n1, int n3, int n5, int Cexp,
                             int Dp, float invdenom)
{
  const int  ch   = n1 + 3 * n3 + 5 * n5;
  const long vol  = (long)Dp * Dp * Dp;
  const long total = (long)Bn * Cexp * vol;
  for (long idx = (long)blockIdx.x * blockDim.x + threadIdx.x; idx < total;
       idx += (long)gridDim.x * blockDim.x) {
    int xpx = (int)(idx % Dp); long t = idx / Dp;
    int ypx = (int)(t % Dp);   t /= Dp;
    int zpx = (int)(t % Dp);   t /= Dp;
    int cc  = (int)(t % Cexp);
    int b   = (int)(t / Cexp);
    const int xi = xpx - 3, yi = ypx - 3, zi = zpx - 3;
    float o = 0.f;
    if (xi >= 0 && xi < 16 && yi >= 0 && yi < 16 && zi >= 0 && zi < 16) {
      const int vox = (zi * 16 + yi) * 16 + xi;
      auto getv = [&](int c) -> float {
        const int f = (c < n1) ? c
                    : (c < n1 + 3 * n3) ? n1 + (c - n1) / 3
                    : n1 + n3 + (c - n1 - 3 * n3) / 5;
        float v = y[((long)b * Cout + c) * 4096 + vox] *
                  rsqrtf(norm2[f] * invdenom + 1e-5f);
        if (c < n1) v = fmaxf(v + bias[c], 0.f);
        return v;
      };
      if (cc < ch) {
        o = getv(cc);
      } else {
        const int j  = cc - ch;
        const int mm = j / 9;
        const int a  = (j % 9) / 3;
        const int bb = j % 3;
        o = getv(n1 + 3 * mm + a) * getv(n1 + 3 * mm + bb);
      }
    }
    xp[idx] = (_Float16)o;
  }
}

// out[b] = mean(conv6) * alpha * 0.1 + sum_c lin_w[c]*sum(x[b,c]) + lin_b
__global__ void final_k(const float* __restrict__ y, const float* __restrict__ x0,
                        const float* __restrict__ lw, const float* __restrict__ lb,
                        const float* __restrict__ alpha, float* __restrict__ out)
{
  const int b = blockIdx.x, tid = threadIdx.x;
  float s1 = 0.f;
  for (int i = tid; i < 4096; i += 256) s1 += y[(long)b * 4096 + i];
  float s2 = 0.f;
  const long per = 5L * 32768;
  for (long i = tid; i < per; i += 256) {
    const int c = (int)(i >> 15);
    s2 += x0[(long)b * per + i] * lw[c];
  }
  __shared__ float sa[256], sb[256];
  sa[tid] = s1; sb[tid] = s2; __syncthreads();
  for (int o = 128; o > 0; o >>= 1) {
    if (tid < o) { sa[tid] += sa[tid + o]; sb[tid] += sb[tid + o]; }
    __syncthreads();
  }
  if (tid == 0) out[b] = sa[0] * (1.f / 4096.f) * alpha[0] * 0.1f + sb[0] + lb[0];
}

// =============================== host ===============================
extern "C" void kernel_launch(void* const* d_in, const int* in_sizes, int n_in,
                              void* d_out, int out_size, void* d_ws, size_t ws_size,
                              hipStream_t stream)
{
  (void)in_sizes; (void)n_in; (void)out_size; (void)ws_size;
  // Layer tables (after tensor-product concat). NCH = ceil(Cin*49/4) padded
  // to a multiple of 8 (zero-filled weights / guarded A rows).
  static const int CIN [6] = {5, 46, 51, 117, 117, 117};
  static const int COUT[6] = {19, 24, 45, 45, 45, 1};
  static const int NT  [6] = {2, 2, 3, 3, 3, 1};
  static const int NCH [6] = {64, 568, 632, 1440, 1440, 1440};
  static const int DP  [6] = {38, 22, 22, 22, 22, 22};
  static const int STR [6] = {2, 1, 1, 1, 1, 1};
  static const int N1  [5] = {10, 10, 16, 16, 16};
  static const int N3  [5] = {3, 3, 8, 8, 8};
  static const int N5  [5] = {0, 1, 1, 1, 1};

  char* ws = (char*)d_ws;
  size_t off = 0;
  auto carve = [&](size_t bytes) -> char* {
    char* p = ws + off;
    off = (off + bytes + 255) & ~(size_t)255;
    return p;
  };
  unsigned*  wB      = (unsigned*) carve((size_t)3 * 1440 * 256 * 4);          // max layer
  int*       rowTab  = (int*)      carve((size_t)5760 * 4);
  _Float16*  xpad    = (_Float16*) carve(((size_t)2 * 117 * 22 * 22 * 22 + 64) * 2);
  float*     convout = (float*)    carve((size_t)2 * 45 * 4096 * 4);
  float*     norm2   = (float*)    carve(64 * 4);

  const float* x0 = (const float*)d_in[0];

  { // layer-0 padded f16 input
    long total = 2L * 5 * 38 * 38 * 38;
    pad_input_k<<<(int)((total + 255) / 256), 256, 0, stream>>>(x0, xpad, 2, 5, 32, 38);
  }

  for (int i = 0; i < 6; ++i) {
    const long wtot = (long)NT[i] * NCH[i] * 256;
    prep_wB_k<<<(int)((wtot + 255) / 256), 256, 0, stream>>>(
        (const float*)d_in[1 + i], wB, COUT[i], CIN[i], NT[i], NCH[i]);

    const int nrows = NCH[i] * 4;
    prep_rows_k<<<(nrows + 255) / 256, 256, 0, stream>>>(rowTab, CIN[i], DP[i], nrows);

    conv_wmma_k<<<2 * NT[i] * 32, 256, 0, stream>>>(
        xpad, wB, rowTab, convout, CIN[i], COUT[i], NT[i], NCH[i], DP[i], STR[i]);

    if (i < 5) {
      zero_k<<<1, 64, 0, stream>>>(norm2, 64);
      norm_reduce_k<<<2 * COUT[i], 256, 0, stream>>>(convout, norm2, COUT[i], N1[i], N3[i]);
      const int  cexp = CIN[i + 1];
      const long etot = 2L * cexp * 22 * 22 * 22;
      expand_pad_k<<<(int)((etot + 255) / 256), 256, 0, stream>>>(
          convout, norm2, (const float*)d_in[7 + i], xpad,
          2, COUT[i], N1[i], N3[i], N5[i], cexp, 22, 1.f / (2.f * 4096.f));
    } else {
      final_k<<<2, 256, 0, stream>>>(convout, x0,
          (const float*)d_in[12], (const float*)d_in[13], (const float*)d_in[14],
          (float*)d_out);
    }
  }
}
